// MambaBlock_33045478376036
// MI455X (gfx1250) — matrix operand order repaired and checked by reading, not compiled, not run on hardware
//
#include <hip/hip_runtime.h>

// ---- problem dims ----
#define DM   1024
#define DI   2048
#define DSN  16
#define DCV  4
#define DRK  64
#define BB   2
#define LL   2048
#define MTOT (BB * LL)   // 4096 token rows
#define NXZ  (2 * DI)    // 4096
#define NDBL 96          // DR + 2*DS
#define NDBL_PAD 128     // x_proj weight rows padded to tile multiple

typedef __attribute__((ext_vector_type(16))) __bf16 v16bf;
typedef __attribute__((ext_vector_type(8)))  __bf16 v8bf;
typedef __attribute__((ext_vector_type(8)))  float  v8f;
typedef int i32x4_ __attribute__((vector_size(16)));   // matches builtin's V4i

// CDNA5 async global->LDS path (gfx1250), with sync fallback
#if defined(__AMDGCN__) && __has_builtin(__builtin_amdgcn_global_load_async_to_lds_b128) && \
    __has_builtin(__builtin_amdgcn_s_wait_asynccnt)
#define USE_ASYNC 1
#define GLD_ASYNC_B128(gsrc, ldst)                                        \
    __builtin_amdgcn_global_load_async_to_lds_b128(                       \
        (__attribute__((address_space(1))) i32x4_*)(void*)(gsrc),         \
        (__attribute__((address_space(3))) i32x4_*)(void*)(ldst), 0, 0)
#define WAIT_STAGE() __builtin_amdgcn_s_wait_asynccnt(0)
#else
#define USE_ASYNC 0
#define WAIT_STAGE()
#endif

__device__ __forceinline__ __bf16 to_bf16(float f) {
    unsigned u = __float_as_uint(f);
    u += 0x7fffu + ((u >> 16) & 1u);          // round-to-nearest-even
    unsigned short s = (unsigned short)(u >> 16);
    __bf16 r;
    __builtin_memcpy(&r, &s, 2);
    return r;
}

__device__ __forceinline__ float sigmoidf_(float x) { return 1.f / (1.f + __expf(-x)); }
__device__ __forceinline__ float softplusf_(float x) { return x > 20.f ? x : log1pf(__expf(x)); }

// ---------------------------------------------------------------------------
// fp32 -> bf16 converter
// ---------------------------------------------------------------------------
__global__ __launch_bounds__(256)
void cvt_f32_to_bf16(const float* __restrict__ in, __bf16* __restrict__ out, int n) {
    int i = blockIdx.x * 256 + threadIdx.x;
    if (i < n) out[i] = to_bf16(in[i]);
}

// convert with zero padding past n_src
__global__ __launch_bounds__(256)
void cvt_pad_bf16(const float* __restrict__ in, __bf16* __restrict__ out,
                  int n_src, int n_dst) {
    int i = blockIdx.x * 256 + threadIdx.x;
    if (i < n_dst) out[i] = (i < n_src) ? to_bf16(in[i]) : to_bf16(0.f);
}

// extract dt_low (first DR cols of x_dbl) as bf16 [MTOT][DRK]
__global__ __launch_bounds__(256)
void extract_dtlow_bf16(const float* __restrict__ xdbl, __bf16* __restrict__ out) {
    int i = blockIdx.x * 256 + threadIdx.x;      // 0 .. MTOT*DRK-1
    int m = i >> 6;
    int r = i & 63;
    out[i] = to_bf16(xdbl[(size_t)m * NDBL + r]);
}

// ---------------------------------------------------------------------------
// WMMA bf16 TN GEMM: C[M,N] = A[M,K] * B[N,K]^T  (row-major, K-contiguous)
// block = 128 threads (4 waves); block tile 64x128, BK=32, double-buffered
// via GLOBAL_LOAD_ASYNC_TO_LDS_B128 (ASYNCcnt) with sync fallback.
// wave tile 32x64 -> 2x4 v_wmma_f32_16x16x32_bf16 per K step per wave.
// LDS rows padded 32->40 halves (80B stride) for conflict-free b128 frags.
// Npad = B rows staged (multiple of 128, zero-padded); N = logical/stride.
// mode 0: C = v ; mode 1: C = softplus(v + bias[n])
// ---------------------------------------------------------------------------
#define LDP 40

__global__ __launch_bounds__(128)
void wmma_gemm_tn(const __bf16* __restrict__ A, const __bf16* __restrict__ B,
                  float* __restrict__ C, int M, int N, int Npad, int K,
                  int mode, const float* __restrict__ bias)
{
    __shared__ __align__(128) __bf16 As[2][64][LDP];
    __shared__ __align__(128) __bf16 Bs[2][128][LDP];

    const int t    = threadIdx.x;
    const int bm   = blockIdx.y * 64;
    const int bn   = blockIdx.x * 128;
    const int w    = t >> 5;
    const int lane = t & 31;
    const int wm   = (w >> 1) * 32;      // waves 2x2 over (64m x 128n)
    const int wn   = (w & 1) * 64;
    const int ln   = lane & 15;          // row/col in 16x16 tile
    const int kh   = lane >> 4;          // K-half selector per ISA layouts

    const int ra = t >> 1;               // A tile: 64 rows, 2 threads/row
    const int ca = (t & 1) * 16;         // 32-byte segment (16 halves)
    const int rb = t;                    // B tile: 128 rows, 1 thread/row

    v8f acc[2][4] = {};

    // stage one (A,B) K-slice into buffer `buf`
    auto stage = [&](int buf, int k0) {
        const __bf16* ga = A + (size_t)(bm + ra) * K + k0 + ca;
        const __bf16* gb = B + (size_t)(bn + rb) * K + k0;
#if USE_ASYNC
        GLD_ASYNC_B128(ga,     &As[buf][ra][ca]);
        GLD_ASYNC_B128(ga + 8, &As[buf][ra][ca + 8]);
#pragma unroll
        for (int c = 0; c < 4; ++c)
            GLD_ASYNC_B128(gb + c * 8, &Bs[buf][rb][c * 8]);
#else
        const uint4* sa = reinterpret_cast<const uint4*>(ga);
        uint4* da = reinterpret_cast<uint4*>(&As[buf][ra][ca]);
        da[0] = sa[0];
        da[1] = sa[1];
        const uint4* sb = reinterpret_cast<const uint4*>(gb);
#pragma unroll
        for (int c = 0; c < 4; ++c)
            *reinterpret_cast<uint4*>(&Bs[buf][rb][c * 8]) = sb[c];
        __builtin_prefetch(ga + 32, 0, 1);
        __builtin_prefetch(gb + 32, 0, 1);
#endif
    };

    stage(0, 0);
    int cur = 0;

    for (int k0 = 0; k0 < K; k0 += 32) {
        WAIT_STAGE();           // this wave's staged async writes landed
        __syncthreads();        // everyone's writes visible
        if (k0 + 32 < K) stage(cur ^ 1, k0 + 32);

        // --- build fragments per CDNA5 16-bit WMMA VGPR layouts ---
        // A 16x32: lane row=ln holds K[kh*8..+8) and K[16+kh*8..+8)
        v16bf afrag[2], bfrag[4];
#pragma unroll
        for (int i = 0; i < 2; ++i) {
            int m = wm + i * 16 + ln;
            const v8bf lo = *reinterpret_cast<const v8bf*>(&As[cur][m][kh * 8]);
            const v8bf hi = *reinterpret_cast<const v8bf*>(&As[cur][m][16 + kh * 8]);
#pragma unroll
            for (int q = 0; q < 8; ++q) { afrag[i][q] = lo[q]; afrag[i][q + 8] = hi[q]; }
        }
        // B 32x16: lane col=ln holds K[kh*8..+8) and K[16+kh*8..+8)
#pragma unroll
        for (int j = 0; j < 4; ++j) {
            int n = wn + j * 16 + ln;
            const v8bf lo = *reinterpret_cast<const v8bf*>(&Bs[cur][n][kh * 8]);
            const v8bf hi = *reinterpret_cast<const v8bf*>(&Bs[cur][n][16 + kh * 8]);
#pragma unroll
            for (int q = 0; q < 8; ++q) { bfrag[j][q] = lo[q]; bfrag[j][q + 8] = hi[q]; }
        }

#pragma unroll
        for (int i = 0; i < 2; ++i)
#pragma unroll
            for (int j = 0; j < 4; ++j)
                acc[i][j] = __builtin_amdgcn_wmma_f32_16x16x32_bf16(
                    false, afrag[i], false, bfrag[j], (short)0, acc[i][j], false, false);

        cur ^= 1;
    }

    // --- epilogue: C 16x16 f32 layout: VGPR rr -> M = kh*8 + rr, N = ln ---
#pragma unroll
    for (int i = 0; i < 2; ++i) {
#pragma unroll
        for (int j = 0; j < 4; ++j) {
            int gn = bn + wn + j * 16 + ln;
            if (gn >= N) continue;
            float bval = (mode == 1) ? bias[gn] : 0.f;
#pragma unroll
            for (int rr = 0; rr < 8; ++rr) {
                int gm = bm + wm + i * 16 + kh * 8 + rr;
                float v = acc[i][j][rr];
                if (mode == 1) v = softplusf_(v + bval);
                C[(size_t)gm * N + gn] = v;
            }
        }
    }
    (void)M; (void)Npad;
}

// ---------------------------------------------------------------------------
// causal depthwise conv1d (DC=4) + bias + SiLU. thread per (b,d), loop over L.
// ---------------------------------------------------------------------------
__global__ __launch_bounds__(256)
void conv_silu_kernel(const float* __restrict__ xz, const float* __restrict__ conv_w,
                      const float* __restrict__ conv_b,
                      float* __restrict__ xconv, __bf16* __restrict__ xbf)
{
    int gid = blockIdx.x * 256 + threadIdx.x;  // 0 .. BB*DI-1
    int b = gid / DI;
    int d = gid % DI;
    float w0 = conv_w[d * DCV + 0], w1 = conv_w[d * DCV + 1];
    float w2 = conv_w[d * DCV + 2], w3 = conv_w[d * DCV + 3];
    float bia = conv_b[d];
    float h0 = 0.f, h1 = 0.f, h2 = 0.f;
    for (int l = 0; l < LL; ++l) {
        size_t m = (size_t)b * LL + l;
        float xv  = xz[m * NXZ + d];
        float acc = w0 * h0 + w1 * h1 + w2 * h2 + w3 * xv + bia;
        h0 = h1; h1 = h2; h2 = xv;
        float s = acc * sigmoidf_(acc);
        xconv[m * DI + d] = s;
        xbf[m * DI + d]   = to_bf16(s);
    }
}

// ---------------------------------------------------------------------------
// selective scan: thread per (b,d); 16-wide state in registers over L steps.
// B/C staged in LDS in 32-step chunks; fuses +x*D skip and *silu(z) gate.
// ---------------------------------------------------------------------------
__global__ __launch_bounds__(256)
void scan_kernel(const float* __restrict__ delta, const float* __restrict__ xconv,
                 const float* __restrict__ xdbl, const float* __restrict__ xz,
                 const float* __restrict__ A_log, const float* __restrict__ Dp,
                 __bf16* __restrict__ ybf)
{
    __shared__ float Bsh[32][16];
    __shared__ float Csh[32][16];
    int t   = threadIdx.x;
    int gid = blockIdx.x * 256 + t;            // 0 .. BB*DI-1
    int b = gid / DI;
    int d = gid % DI;

    float a[16], h[16];
#pragma unroll
    for (int n = 0; n < 16; ++n) {
        a[n] = -__expf(A_log[(size_t)d * DSN + n]);
        h[n] = 0.f;
    }
    float Dv = Dp[d];

    for (int l0 = 0; l0 < LL; l0 += 32) {
#pragma unroll
        for (int i = 0; i < 4; ++i) {
            int idx = i * 256 + t;             // 0..1023
            int s = idx >> 5;
            int c = idx & 31;
            float v = xdbl[((size_t)b * LL + l0 + s) * NDBL + DRK + c];
            if (c < 16) Bsh[s][c] = v; else Csh[s][c - 16] = v;
        }
        __syncthreads();

        for (int s = 0; s < 32; ++s) {
            size_t m  = (size_t)b * LL + l0 + s;
            float dlt = delta[m * DI + d];
            float xv  = xconv[m * DI + d];
            float dx  = dlt * xv;
            float y   = 0.f;
#pragma unroll
            for (int n = 0; n < 16; ++n) {
                float e = __expf(dlt * a[n]);
                h[n] = e * h[n] + dx * Bsh[s][n];
                y += h[n] * Csh[s][n];
            }
            y += xv * Dv;
            float zv = xz[m * NXZ + DI + d];
            y *= zv * sigmoidf_(zv);
            ybf[m * DI + d] = to_bf16(y);
        }
        __syncthreads();
    }
}

// ---------------------------------------------------------------------------
extern "C" void kernel_launch(void* const* d_in, const int* in_sizes, int n_in,
                              void* d_out, int out_size, void* d_ws, size_t ws_size,
                              hipStream_t stream)
{
    (void)in_sizes; (void)n_in; (void)out_size; (void)ws_size;
    const float* hs         = (const float*)d_in[0];
    const float* in_proj_w  = (const float*)d_in[1];
    const float* conv_w     = (const float*)d_in[2];
    const float* conv_b     = (const float*)d_in[3];
    const float* x_proj_w   = (const float*)d_in[4];
    const float* dt_proj_w  = (const float*)d_in[5];
    const float* dt_proj_b  = (const float*)d_in[6];
    const float* A_log      = (const float*)d_in[7];
    const float* Dp         = (const float*)d_in[8];
    const float* out_proj_w = (const float*)d_in[9];
    float* out = (float*)d_out;

    char* p = (char*)d_ws;
    auto alloc = [&](size_t bytes) {
        char* r = p;
        p += (bytes + 255) & ~(size_t)255;
        return r;
    };

    __bf16* hs_bf      = (__bf16*)alloc((size_t)MTOT * DM * 2);
    __bf16* inw_bf     = (__bf16*)alloc((size_t)NXZ * DM * 2);
    float*  xz_f       = (float*)alloc((size_t)MTOT * NXZ * 4);
    float*  xconv_f    = (float*)alloc((size_t)MTOT * DI * 4);
    __bf16* x_bf       = (__bf16*)alloc((size_t)MTOT * DI * 2);
    __bf16* xprojw_bf  = (__bf16*)alloc((size_t)NDBL_PAD * DI * 2);
    float*  xdbl_f     = (float*)alloc((size_t)MTOT * NDBL * 4);
    __bf16* dtlow_bf   = (__bf16*)alloc((size_t)MTOT * DRK * 2);
    __bf16* dtprojw_bf = (__bf16*)alloc((size_t)DI * DRK * 2);
    float*  delta_f    = (float*)alloc((size_t)MTOT * DI * 4);
    __bf16* outw_bf    = (__bf16*)alloc((size_t)DM * DI * 2);
    __bf16* y_bf       = (__bf16*)alloc((size_t)MTOT * DI * 2);

    auto cvt = [&](const float* src, __bf16* dst, int n) {
        cvt_f32_to_bf16<<<(n + 255) / 256, 256, 0, stream>>>(src, dst, n);
    };

    // 0) bf16 copies of activations + all GEMM weights
    cvt(hs, hs_bf, MTOT * DM);
    cvt(in_proj_w, inw_bf, NXZ * DM);
    cvt_pad_bf16<<<(NDBL_PAD * DI) / 256, 256, 0, stream>>>(
        x_proj_w, xprojw_bf, NDBL * DI, NDBL_PAD * DI);
    cvt(dt_proj_w, dtprojw_bf, DI * DRK);
    cvt(out_proj_w, outw_bf, DM * DI);

    // 1) in_proj: xz[M,4096] = hs[M,1024] x in_proj_w[4096,1024]^T
    wmma_gemm_tn<<<dim3(NXZ / 128, MTOT / 64), 128, 0, stream>>>(
        hs_bf, inw_bf, xz_f, MTOT, NXZ, NXZ, DM, 0, nullptr);

    // 2) causal depthwise conv + SiLU on x half
    conv_silu_kernel<<<(BB * DI) / 256, 256, 0, stream>>>(
        xz_f, conv_w, conv_b, xconv_f, x_bf);

    // 3) x_proj: x_dbl[M,96] = x[M,2048] x x_proj_w[96(pad128),2048]^T
    wmma_gemm_tn<<<dim3(NDBL_PAD / 128, MTOT / 64), 128, 0, stream>>>(
        x_bf, xprojw_bf, xdbl_f, MTOT, NDBL, NDBL_PAD, DI, 0, nullptr);

    // 4) slice dt_low -> bf16
    extract_dtlow_bf16<<<(MTOT * DRK) / 256, 256, 0, stream>>>(xdbl_f, dtlow_bf);

    // 5) dt_proj + softplus bias: delta[M,2048]
    wmma_gemm_tn<<<dim3(DI / 128, MTOT / 64), 128, 0, stream>>>(
        dtlow_bf, dtprojw_bf, delta_f, MTOT, DI, DI, DRK, 1, dt_proj_b);

    // 6) selective scan + skip + gate -> y (bf16)
    scan_kernel<<<(BB * DI) / 256, 256, 0, stream>>>(
        delta_f, xconv_f, xdbl_f, xz_f, A_log, Dp, y_bf);

    // 7) out_proj: out[M,1024] = y[M,2048] x out_proj_w[1024,2048]^T
    wmma_gemm_tn<<<dim3(DM / 128, MTOT / 64), 128, 0, stream>>>(
        y_bf, outw_bf, out, MTOT, DM, DM, DI, 0, nullptr);
}
